// TopKRouter_14998025797639
// MI455X (gfx1250) — matrix operand order repaired and checked by reading, not compiled, not run on hardware
//
#include <hip/hip_runtime.h>

// ---------------------------------------------------------------------------
// MoE Top-K router for MI455X (gfx1250, wave32, WMMA).
//
// logits = x(16384x2048) @ W_gate^T(2048x64), softmax, top-2 (+renorm), aux.
// Roofline: x stream = 134 MB @ 23.3 TB/s ~ 5.8us floor; needs ~750 TFLOP/s
// matrix rate -> v_wmma_f32_16x16x32_bf16. bf16 packing via v_perm_b32;
// W staged once per block into LDS (256 KB) pre-packed in WMMA-B layout.
// Epilogue: branchless top-2 via order-preserving u32 keys (idx in low 6
// bits) + v_permlane16_b32 xor-reductions -- no EXEC-mask divergence.
// ---------------------------------------------------------------------------

typedef __attribute__((ext_vector_type(16))) __bf16        v16bf;
typedef __attribute__((ext_vector_type(8)))  float         v8f;
typedef __attribute__((ext_vector_type(8)))  unsigned int  v8u;

#define N_EXPERTS        64
#define DIM              2048
#define N_TOKENS         16384
#define TOKENS_PER_WAVE  32
#define WAVES_PER_BLOCK  8
#define TOKENS_PER_BLOCK (TOKENS_PER_WAVE * WAVES_PER_BLOCK)  // 256
#define WBUF_BYTES       (N_EXPERTS * DIM * 2)                // 256 KB bf16 W

// permlane16 nibble selectors for lane -> lane ^ s within each 16-lane row
#define SEL1_LO 0x67452301u
#define SEL1_HI 0xefcdab89u
#define SEL2_LO 0x54761032u
#define SEL2_HI 0xdcfe98bau
#define SEL4_LO 0x32107654u
#define SEL4_HI 0xba98fedcu
#define SEL8_LO 0xfedcba98u
#define SEL8_HI 0x76543210u

__device__ __forceinline__ unsigned permx_u(unsigned v, unsigned lo, unsigned hi) {
    return __builtin_amdgcn_permlane16(v, v, lo, hi, false, false);
}
__device__ __forceinline__ float permx_f(float v, unsigned lo, unsigned hi) {
    return __uint_as_float(permx_u(__float_as_uint(v), lo, hi));
}
__device__ __forceinline__ unsigned umax2(unsigned a, unsigned b) {
    return a > b ? a : b;
}
// order-preserving float-bits -> unsigned transform and inverse
__device__ __forceinline__ unsigned mono(float f) {
    unsigned u = __float_as_uint(f);
    return u ^ (unsigned)(((int)u >> 31) | (int)0x80000000);
}
__device__ __forceinline__ float unmono(unsigned k) {
    unsigned nk = ~k;
    return __uint_as_float(k ^ (unsigned)(((int)nk >> 31) | (int)0x80000000));
}

// result = (hi16(b) << 16) | hi16(a)  : pack 2 floats to 2 bf16 (truncate)
__device__ __forceinline__ unsigned int pack2(unsigned int a, unsigned int b) {
    return __builtin_amdgcn_perm(b, a, 0x07060302u);
}

// A-matrix 16x32 bf16 tile (ISA 7.12.2): lanes 0-15 (M=0..15) hold
// K = {0..7} in v0-3 and {16..23} in v4-7; lanes 16-31 hold K = {8..15},{24..31}.
__device__ __forceinline__ v16bf load_tile_A(const float* __restrict__ row,
                                             int kb, int half) {
    const uint4* p0 = (const uint4*)(row + kb + 8 * half);
    const uint4* p1 = (const uint4*)(row + kb + 16 + 8 * half);
    uint4 f0 = p0[0], f1 = p0[1], f2 = p1[0], f3 = p1[1];
    v8u u;
    u[0] = pack2(f0.x, f0.y); u[1] = pack2(f0.z, f0.w);
    u[2] = pack2(f1.x, f1.y); u[3] = pack2(f1.z, f1.w);
    u[4] = pack2(f2.x, f2.y); u[5] = pack2(f2.z, f2.w);
    u[6] = pack2(f3.x, f3.y); u[7] = pack2(f3.z, f3.w);
    return __builtin_bit_cast(v16bf, u);
}

__global__ __launch_bounds__(256)
void router_gemm_topk_kernel(const float* __restrict__ x,
                             const float* __restrict__ W,
                             float* __restrict__ out_idx,   // [N_TOKENS*2]
                             float* __restrict__ out_w,     // [N_TOKENS*2]
                             float* __restrict__ g_probsum, // [64]
                             float* __restrict__ g_count) { // [64]
    extern __shared__ unsigned char smem[];
    unsigned char* wbuf        = smem;                         // 256 KB
    float*         lds_probsum = (float*)(smem + WBUF_BYTES);  // 64
    float*         lds_count   = lds_probsum + N_EXPERTS;      // 64

    const int tid = threadIdx.x;
    if (tid < N_EXPERTS) { lds_probsum[tid] = 0.0f; lds_count[tid] = 0.0f; }

    // ---- Stage W (f32 global) -> bf16 LDS, already in WMMA-B layout. ------
    // B-matrix 32x16 tile: lanes 0-15 = cols 0-15 / K 0..7,16..23 (2 per dword),
    // lanes 16-31 = same cols / K 8..15,24..31.  LDS layout:
    //   [kb(64)][tile(4)][lane(32)][8 dwords]  (32 B per lane per tile)
#pragma unroll 4
    for (int i = 0; i < 64; ++i) {
        const int u = i * 256 + tid;
        const int e = u >> 8;            // expert 0..63
        const int k = (u & 255) << 3;    // k multiple of 8
        const uint4* wp = (const uint4*)(W + (size_t)e * DIM + k);
        uint4 f0 = wp[0], f1 = wp[1];
        const int kb   = k >> 5;
        const int kk   = k & 31;
        const int hf   = (kk >> 3) & 1;
        const int v    = (kk >> 4) << 2;           // 0 or 4
        const int lane = (e & 15) + (hf << 4);
        const int t    = e >> 4;
        uint4 d;
        d.x = pack2(f0.x, f0.y); d.y = pack2(f0.z, f0.w);
        d.z = pack2(f1.x, f1.y); d.w = pack2(f1.z, f1.w);
        *(uint4*)(wbuf + ((((kb * 4 + t) * 32 + lane) * 32) + v * 4)) = d;
    }
    __syncthreads();

    // ---- GEMM: 32 tokens per wave (2 A tiles), 4 expert tiles. ------------
    const int wave = tid >> 5;
    const int lane = tid & 31;
    const int half = lane >> 4;
    const int col  = lane & 15;
    const int token_base = blockIdx.x * TOKENS_PER_BLOCK + wave * TOKENS_PER_WAVE;

    const float* xrow0 = x + (size_t)(token_base + col) * DIM;
    const float* xrow1 = xrow0 + (size_t)16 * DIM;

    v8f acc0[4] = {};
    v8f acc1[4] = {};

    for (int kb = 0; kb < DIM; kb += 32) {
        v16bf a0 = load_tile_A(xrow0, kb, half);
        v16bf a1 = load_tile_A(xrow1, kb, half);
        const v8u* bbase =
            (const v8u*)(wbuf + (((kb >> 5) * 4) * 32 + lane) * 32);
#pragma unroll
        for (int t = 0; t < 4; ++t) {
            v16bf b = __builtin_bit_cast(v16bf, bbase[t * 32]);
            acc0[t] = __builtin_amdgcn_wmma_f32_16x16x32_bf16(
                false, a0, false, b, (short)0, acc0[t], false, false);
            acc1[t] = __builtin_amdgcn_wmma_f32_16x16x32_bf16(
                false, a1, false, b, (short)0, acc1[t], false, false);
        }
    }

    // ---- Epilogue: branchless key-based top-2 + softmax (permlane16). -----
    // C/D layout: VGPR r = row M=r (lanes 0-15) / M=r+8 (lanes 16-31),
    // column N = lane%16.  Lane's experts: e_t = 16t + col.
    // Key = (mono(logit) & ~63) | (63 - expert): unsigned max picks the
    // largest logit, ties resolve to the lower expert index (JAX top_k rule).
    float psum_local[4] = {0.0f, 0.0f, 0.0f, 0.0f};
    const unsigned rv0 = 63 - col, rv1 = 47 - col, rv2 = 31 - col, rv3 = 15 - col;

    auto process16 = [&](v8f (&acc)[4], int tbase) {
#pragma unroll
        for (int r = 0; r < 8; ++r) {
            const int token = tbase + r + 8 * half;
            float v0 = acc[0][r], v1 = acc[1][r], v2 = acc[2][r], v3 = acc[3][r];

            unsigned k0 = (mono(v0) & ~63u) | rv0;
            unsigned k1 = (mono(v1) & ~63u) | rv1;
            unsigned k2 = (mono(v2) & ~63u) | rv2;
            unsigned k3 = (mono(v3) & ~63u) | rv3;

            // top-1 key (row max)
            unsigned kmax = umax2(umax2(k0, k1), umax2(k2, k3));
            kmax = umax2(kmax, permx_u(kmax, SEL1_LO, SEL1_HI));
            kmax = umax2(kmax, permx_u(kmax, SEL2_LO, SEL2_HI));
            kmax = umax2(kmax, permx_u(kmax, SEL4_LO, SEL4_HI));
            kmax = umax2(kmax, permx_u(kmax, SEL8_LO, SEL8_HI));
            const int   i0 = 63 - (int)(kmax & 63u);
            const float m  = unmono(kmax);

            // top-2: kill the unique winner, reduce max again
            unsigned z0 = (k0 == kmax) ? 0u : k0;
            unsigned z1 = (k1 == kmax) ? 0u : k1;
            unsigned z2 = (k2 == kmax) ? 0u : k2;
            unsigned z3 = (k3 == kmax) ? 0u : k3;
            unsigned ksec = umax2(umax2(z0, z1), umax2(z2, z3));
            ksec = umax2(ksec, permx_u(ksec, SEL1_LO, SEL1_HI));
            ksec = umax2(ksec, permx_u(ksec, SEL2_LO, SEL2_HI));
            ksec = umax2(ksec, permx_u(ksec, SEL4_LO, SEL4_HI));
            ksec = umax2(ksec, permx_u(ksec, SEL8_LO, SEL8_HI));
            const int   i1  = 63 - (int)(ksec & 63u);
            const float b1f = unmono(ksec);

            // softmax denominator
            float e0 = __expf(v0 - m), e1 = __expf(v1 - m);
            float e2 = __expf(v2 - m), e3 = __expf(v3 - m);
            float sum = (e0 + e1) + (e2 + e3);
            sum += permx_f(sum, SEL1_LO, SEL1_HI);
            sum += permx_f(sum, SEL2_LO, SEL2_HI);
            sum += permx_f(sum, SEL4_LO, SEL4_HI);
            sum += permx_f(sum, SEL8_LO, SEL8_HI);
            const float inv = 1.0f / sum;

            psum_local[0] += e0 * inv; psum_local[1] += e1 * inv;
            psum_local[2] += e2 * inv; psum_local[3] += e3 * inv;

            const float p0 = inv;                  // exp(m - m) == 1
            const float p1 = __expf(b1f - m) * inv;
            const float dn = p0 + p1 + 1e-8f;

            if (col == 0) {
                *(float2*)(out_idx + 2 * token) =
                    make_float2((float)i0, (float)i1);
                *(float2*)(out_w + 2 * token) =
                    make_float2(p0 / dn, p1 / dn);
                atomicAdd(&lds_count[i0], 1.0f);
                atomicAdd(&lds_count[i1], 1.0f);
            }
        }
    };
    process16(acc0, token_base);
    process16(acc1, token_base + 16);

#pragma unroll
    for (int t = 0; t < 4; ++t)
        atomicAdd(&lds_probsum[t * 16 + col], psum_local[t]);

    __syncthreads();
    if (tid < N_EXPERTS) {
        atomicAdd(&g_probsum[tid], lds_probsum[tid]);
        atomicAdd(&g_count[tid],   lds_count[tid]);
    }
}

__global__ void router_aux_loss_kernel(const float* __restrict__ g_probsum,
                                       const float* __restrict__ g_count,
                                       float* __restrict__ aux_out) {
    __shared__ float s[N_EXPERTS];
    const int e = threadIdx.x;
    const float f = g_count[e] * (1.0f / (float)(N_TOKENS * 2));
    const float P = g_probsum[e] * (1.0f / (float)N_TOKENS);
    s[e] = f * P;
    __syncthreads();
    if (e < 32) s[e] += s[e + 32];
    __syncthreads();
    if (e == 0) {
        float t = 0.0f;
#pragma unroll
        for (int i = 0; i < 32; ++i) t += s[i];
        aux_out[0] = (float)N_EXPERTS * t;
    }
}

extern "C" void kernel_launch(void* const* d_in, const int* in_sizes, int n_in,
                              void* d_out, int out_size, void* d_ws, size_t ws_size,
                              hipStream_t stream) {
    (void)in_sizes; (void)n_in; (void)out_size; (void)ws_size;

    const float* x = (const float*)d_in[0];  // (4,4096,2048) f32
    const float* W = (const float*)d_in[1];  // (64,2048) f32

    float* out     = (float*)d_out;
    float* out_idx = out;                          // 16384*2 indices (as float)
    float* out_w   = out + (size_t)N_TOKENS * 2;   // 16384*2 weights
    float* aux     = out + (size_t)N_TOKENS * 4;   // 1 scalar

    float* g_probsum = (float*)d_ws;               // 64 floats
    float* g_count   = g_probsum + N_EXPERTS;      // 64 floats

    hipMemsetAsync(d_ws, 0, 2 * N_EXPERTS * sizeof(float), stream);

    const size_t shmem = WBUF_BYTES + 2 * N_EXPERTS * sizeof(float);
    dim3 grid(N_TOKENS / TOKENS_PER_BLOCK);  // 64 blocks
    router_gemm_topk_kernel<<<grid, 256, shmem, stream>>>(
        x, W, out_idx, out_w, g_probsum, g_count);

    router_aux_loss_kernel<<<1, N_EXPERTS, 0, stream>>>(g_probsum, g_count, aux);
}
